// GRU_19559281066318
// MI455X (gfx1250) — compile-verified, matrix operand order checked by reading, not run on hardware
//
#include <hip/hip_runtime.h>

// ---------------- types ----------------
typedef __bf16 bf16_t;
typedef __attribute__((ext_vector_type(16))) __bf16 v16bf;
typedef __attribute__((ext_vector_type(8)))  __bf16 v8bf;
typedef __attribute__((ext_vector_type(8)))  float  v8f;
typedef __attribute__((ext_vector_type(4)))  float  v4f;

#define TT   512
#define BB   512
#define DH   512
#define DOUT 256
#define NTOT 1536           // 3 gates * DH
#define HAS  520            // padded bf16 LDS row stride (bank-conflict-free b128 frag loads)
#define HFS  514            // padded f32 LDS row stride (half-wave bank split)

// ---------------- WMMA helpers ----------------
__device__ __forceinline__ v16bf cat16(v8bf lo, v8bf hi) {
  return __builtin_shufflevector(lo, hi, 0,1,2,3,4,5,6,7,8,9,10,11,12,13,14,15);
}
__device__ __forceinline__ v16bf load_frag16(const bf16_t* p) {
  v8bf lo = *(const v8bf*)p;          // K = kb + half*8 + 0..7
  v8bf hi = *(const v8bf*)(p + 16);   // K = kb + 16 + half*8 + 0..7
  return cat16(lo, hi);
}
// A fragment: 16 x 32 tile from bf16 row-major (row stride in elements)
__device__ __forceinline__ v16bf load_a_bf16(const bf16_t* base, int stride, int kb) {
  int lane = threadIdx.x & 31;
  return load_frag16(base + (lane & 15) * stride + kb + (lane >> 4) * 8);
}
// B fragment: 32 x 16 tile from pre-transposed weights Wt[n][k], K stride 512
__device__ __forceinline__ v16bf load_b_w(const bf16_t* __restrict__ Wt, int n0, int kb) {
  int lane = threadIdx.x & 31;
  return load_frag16(Wt + (size_t)(n0 + (lane & 15)) * DH + kb + (lane >> 4) * 8);
}
// A fragment from f32 row-major with on-the-fly bf16 conversion
__device__ __forceinline__ v16bf load_a_f32(const float* __restrict__ base, int stride, int kb) {
  int lane = threadIdx.x & 31;
  const float* p = base + (size_t)(lane & 15) * stride + kb + (lane >> 4) * 8;
  v4f x0 = *(const v4f*)p;
  v4f x1 = *(const v4f*)(p + 4);
  v4f x2 = *(const v4f*)(p + 16);
  v4f x3 = *(const v4f*)(p + 20);
  v16bf r;
#pragma unroll
  for (int i = 0; i < 4; ++i) {
    r[i]      = (bf16_t)x0[i];
    r[4 + i]  = (bf16_t)x1[i];
    r[8 + i]  = (bf16_t)x2[i];
    r[12 + i] = (bf16_t)x3[i];
  }
  return r;
}
__device__ __forceinline__ v8f wmma_bf16(v16bf a, v16bf b, v8f c) {
  return __builtin_amdgcn_wmma_f32_16x16x32_bf16(false, a, false, b, (short)0, c, false, false);
}
__device__ __forceinline__ float sigmoid_fast(float x) {
  return 1.f / (1.f + __expf(-x));
}
__device__ __forceinline__ float tanh_fast(float x) {
  x = fminf(12.f, fmaxf(-12.f, x));
  float e = __expf(-2.f * x);
  return (1.f - e) / (1.f + e);
}

// ---------------- prep: transpose + bf16-convert weights, concat biases ----------------
__global__ __launch_bounds__(256) void prep_kernel(
    const float* __restrict__ Wxr, const float* __restrict__ Wxz, const float* __restrict__ Wxh,
    const float* __restrict__ Whr, const float* __restrict__ Whz, const float* __restrict__ Whh,
    const float* __restrict__ Why,
    const float* __restrict__ br, const float* __restrict__ bz, const float* __restrict__ bh,
    bf16_t* __restrict__ Wx_t, bf16_t* __restrict__ Whr_t, bf16_t* __restrict__ Whz_t,
    bf16_t* __restrict__ Whh_t, bf16_t* __restrict__ Why_t, float* __restrict__ bcat)
{
  int id = blockIdx.x * blockDim.x + threadIdx.x;       // covers 512*512 exactly
  int k = id >> 9, n = id & 511;                        // source [k][n]
  size_t td = (size_t)n * DH + k;                       // dest  [n][k]
  Wx_t[td]                = (bf16_t)Wxr[id];
  Wx_t[(size_t)DH*DH + td]   = (bf16_t)Wxz[id];
  Wx_t[(size_t)2*DH*DH + td] = (bf16_t)Wxh[id];
  Whr_t[td] = (bf16_t)Whr[id];
  Whz_t[td] = (bf16_t)Whz[id];
  Whh_t[td] = (bf16_t)Whh[id];
  if (n < DOUT) Why_t[(size_t)n * DH + k] = (bf16_t)Why[(size_t)k * DOUT + n];
  if (id < DH) { bcat[id] = br[id]; bcat[DH + id] = bz[id]; bcat[2*DH + id] = bh[id]; }
}

// ---------------- phase 1: Gx = x @ [Wxr|Wxz|Wxh] + bias  (bf16 out) ----------------
// grid (NTOT/128, Mtot/128), 8 waves/WG, each wave: 16x128 tile (8 accumulators)
__global__ __launch_bounds__(256) void gx_kernel(
    const float* __restrict__ x, const bf16_t* __restrict__ Wx_t,
    const float* __restrict__ bcat, bf16_t* __restrict__ Gx)
{
  const int wave = threadIdx.x >> 5, lane = threadIdx.x & 31;
  const int m16 = lane & 15, hsel = lane >> 4;
  const size_t m0 = ((size_t)blockIdx.y * 8 + wave) * 16;
  const int n0 = blockIdx.x * 128;
  v8f acc[8];
#pragma unroll
  for (int a = 0; a < 8; ++a) acc[a] = (v8f){0,0,0,0,0,0,0,0};
  const float* xrow = x + m0 * DH;
  for (int kb = 0; kb < DH; kb += 32) {
    v16bf aX = load_a_f32(xrow, DH, kb);
#pragma unroll
    for (int a = 0; a < 8; ++a)
      acc[a] = wmma_bf16(aX, load_b_w(Wx_t, n0 + a * 16, kb), acc[a]);
  }
#pragma unroll
  for (int a = 0; a < 8; ++a) {
    int col = n0 + a * 16 + m16;
    float b = bcat[col];
#pragma unroll
    for (int i = 0; i < 8; ++i) {
      int mrow = i + 8 * hsel;
      Gx[(m0 + mrow) * NTOT + col] = (bf16_t)(acc[a][i] + b);
    }
  }
}

// ---------------- phase 2: persistent GRU recurrence ----------------
// 32 workgroups; WG owns 16 batch rows; 8 waves, wave w covers gate cols [w*64, w*64+64)
template <bool USE_GX>
__global__ __launch_bounds__(256, 1) void gru_kernel(
    const float* __restrict__ x,
    const bf16_t* __restrict__ Wx_t,
    const bf16_t* __restrict__ Whr_t, const bf16_t* __restrict__ Whz_t,
    const bf16_t* __restrict__ Whh_t, const bf16_t* __restrict__ Why_t,
    const float* __restrict__ bcat, const bf16_t* __restrict__ Gx,
    const float* __restrict__ b_y, float* __restrict__ out)
{
  __shared__ bf16_t hA[16 * HAS];   // bf16 "A operand" buffer: holds h, then r*h, then h_new
  __shared__ float  hF[16 * HFS];   // f32 master copy of h

  const int tid = threadIdx.x;
  const int wave = tid >> 5, lane = tid & 31;
  const int m16 = lane & 15, hsel = lane >> 4;
  const int base = blockIdx.x * 16;     // batch rows owned
  const int nb = wave * 64;             // gate-column slice

  for (int i = tid; i < 16 * HAS; i += 256) hA[i] = (bf16_t)0.0f;
  for (int i = tid; i < 16 * HFS; i += 256) hF[i] = 0.0f;
  __syncthreads();

#pragma unroll 1
  for (int t = 0; t < TT; ++t) {
    const float*  xrow = x + ((size_t)t * BB + base) * DH;
    const bf16_t* gxrow = USE_GX ? (Gx + ((size_t)t * BB + base) * NTOT) : nullptr;

    // ---- Phase A: r, z gates ----
    v8f accr[4], accz[4];
#pragma unroll
    for (int a = 0; a < 4; ++a) { accr[a] = (v8f){0,0,0,0,0,0,0,0}; accz[a] = (v8f){0,0,0,0,0,0,0,0}; }
    for (int kb = 0; kb < DH; kb += 32) {
      v16bf aH = load_a_bf16(hA, HAS, kb);
      v16bf aX;
      if (!USE_GX) aX = load_a_f32(xrow, DH, kb);
#pragma unroll
      for (int a = 0; a < 4; ++a) {
        int n0 = nb + a * 16;
        accr[a] = wmma_bf16(aH, load_b_w(Whr_t, n0, kb), accr[a]);
        accz[a] = wmma_bf16(aH, load_b_w(Whz_t, n0, kb), accz[a]);
        if (!USE_GX) {
          accr[a] = wmma_bf16(aX, load_b_w(Wx_t, n0, kb), accr[a]);
          accz[a] = wmma_bf16(aX, load_b_w(Wx_t, DH + n0, kb), accz[a]);
        }
      }
    }
    float rhv[4][8];
#pragma unroll
    for (int a = 0; a < 4; ++a) {
      int col = nb + a * 16 + m16;
#pragma unroll
      for (int i = 0; i < 8; ++i) {
        int mrow = i + 8 * hsel;
        float pr = accr[a][i], pz = accz[a][i];
        if (USE_GX) {
          const bf16_t* gp = gxrow + (size_t)mrow * NTOT;
          pr += (float)gp[col];
          pz += (float)gp[DH + col];
        } else {
          pr += bcat[col];
          pz += bcat[DH + col];
        }
        float r = sigmoid_fast(pr);
        accz[a][i] = sigmoid_fast(pz);                    // keep z
        rhv[a][i] = r * hF[mrow * HFS + col];             // r * h
      }
    }
    __syncthreads();                                      // all done reading hA as h
#pragma unroll
    for (int a = 0; a < 4; ++a) {
      int col = nb + a * 16 + m16;
#pragma unroll
      for (int i = 0; i < 8; ++i) hA[(i + 8 * hsel) * HAS + col] = (bf16_t)rhv[a][i];
    }
    __syncthreads();                                      // hA now holds r*h

    // ---- Phase B: h_hat, h_new ----
    v8f acch[4];
#pragma unroll
    for (int a = 0; a < 4; ++a) acch[a] = (v8f){0,0,0,0,0,0,0,0};
    for (int kb = 0; kb < DH; kb += 32) {
      v16bf aR = load_a_bf16(hA, HAS, kb);
      v16bf aX;
      if (!USE_GX) aX = load_a_f32(xrow, DH, kb);
#pragma unroll
      for (int a = 0; a < 4; ++a) {
        int n0 = nb + a * 16;
        acch[a] = wmma_bf16(aR, load_b_w(Whh_t, n0, kb), acch[a]);
        if (!USE_GX) acch[a] = wmma_bf16(aX, load_b_w(Wx_t, 2 * DH + n0, kb), acch[a]);
      }
    }
    float hnv[4][8];
#pragma unroll
    for (int a = 0; a < 4; ++a) {
      int col = nb + a * 16 + m16;
#pragma unroll
      for (int i = 0; i < 8; ++i) {
        int mrow = i + 8 * hsel;
        float pre = acch[a][i];
        if (USE_GX) pre += (float)(gxrow + (size_t)mrow * NTOT)[2 * DH + col];
        else        pre += bcat[2 * DH + col];
        float hh = tanh_fast(pre);
        float z = accz[a][i];
        float h = hF[mrow * HFS + col];
        hnv[a][i] = (1.f - z) * h + z * hh;
      }
    }
    __syncthreads();                                      // all done reading hA as r*h
#pragma unroll
    for (int a = 0; a < 4; ++a) {
      int col = nb + a * 16 + m16;
#pragma unroll
      for (int i = 0; i < 8; ++i) {
        int mrow = i + 8 * hsel;
        hF[mrow * HFS + col] = hnv[a][i];
        hA[mrow * HAS + col] = (bf16_t)hnv[a][i];
      }
    }
    __syncthreads();                                      // hA now holds h_new

    // ---- Phase Y: y = h_new @ W_hy + b_y ----
    {
      int ny = wave * 32;
      v8f accy[2] = {(v8f){0,0,0,0,0,0,0,0}, (v8f){0,0,0,0,0,0,0,0}};
      for (int kb = 0; kb < DH; kb += 32) {
        v16bf aH = load_a_bf16(hA, HAS, kb);
        accy[0] = wmma_bf16(aH, load_b_w(Why_t, ny, kb), accy[0]);
        accy[1] = wmma_bf16(aH, load_b_w(Why_t, ny + 16, kb), accy[1]);
      }
#pragma unroll
      for (int a = 0; a < 2; ++a) {
        int col = ny + a * 16 + m16;
        float by = b_y[col];
#pragma unroll
        for (int i = 0; i < 8; ++i) {
          int mrow = i + 8 * hsel;
          out[((size_t)t * BB + base + mrow) * DOUT + col] = accy[a][i] + by;
        }
      }
    }
    // no barrier needed: next phase-A only reads hA; writes are fenced by its barrier
  }

  // ---- final hidden state ----
  for (int i = tid; i < 16 * DH; i += 256) {
    int m = i >> 9, n = i & 511;
    out[(size_t)TT * BB * DOUT + (size_t)(base + m) * DH + n] = hF[m * HFS + n];
  }
}

// ---------------- host launcher ----------------
extern "C" void kernel_launch(void* const* d_in, const int* in_sizes, int n_in,
                              void* d_out, int out_size, void* d_ws, size_t ws_size,
                              hipStream_t stream) {
  (void)in_sizes; (void)n_in; (void)out_size;
  const float* x   = (const float*)d_in[0];
  const float* Wxr = (const float*)d_in[1];
  const float* Whr = (const float*)d_in[2];
  const float* br  = (const float*)d_in[3];
  const float* Wxz = (const float*)d_in[4];
  const float* Whz = (const float*)d_in[5];
  const float* bz  = (const float*)d_in[6];
  const float* Wxh = (const float*)d_in[7];
  const float* Whh = (const float*)d_in[8];
  const float* bh  = (const float*)d_in[9];
  const float* Why = (const float*)d_in[10];
  const float* by  = (const float*)d_in[11];
  float* out = (float*)d_out;

  char* ws = (char*)d_ws;
  size_t off = 0;
  auto alloc = [&](size_t bytes) -> void* {
    void* p = ws + off;
    off += (bytes + 255) & ~(size_t)255;
    return p;
  };
  bf16_t* Wx_t  = (bf16_t*)alloc((size_t)NTOT * DH * 2);
  bf16_t* Whr_t = (bf16_t*)alloc((size_t)DH * DH * 2);
  bf16_t* Whz_t = (bf16_t*)alloc((size_t)DH * DH * 2);
  bf16_t* Whh_t = (bf16_t*)alloc((size_t)DH * DH * 2);
  bf16_t* Why_t = (bf16_t*)alloc((size_t)DOUT * DH * 2);
  float*  bcat  = (float*)alloc((size_t)NTOT * 4);
  const size_t gx_bytes = (size_t)TT * BB * NTOT * 2;   // 768 MB
  const bool use_gx = (ws_size >= off + gx_bytes);
  bf16_t* Gx = use_gx ? (bf16_t*)alloc(gx_bytes) : nullptr;

  prep_kernel<<<(DH * DH) / 256, 256, 0, stream>>>(
      Wxr, Wxz, Wxh, Whr, Whz, Whh, Why, br, bz, bh,
      Wx_t, Whr_t, Whz_t, Whh_t, Why_t, bcat);

  if (use_gx) {
    gx_kernel<<<dim3(NTOT / 128, (TT * BB) / 128), 256, 0, stream>>>(x, Wx_t, bcat, Gx);
    gru_kernel<true><<<BB / 16, 256, 0, stream>>>(
        x, Wx_t, Whr_t, Whz_t, Whh_t, Why_t, bcat, Gx, by, out);
  } else {
    gru_kernel<false><<<BB / 16, 256, 0, stream>>>(
        x, Wx_t, Whr_t, Whz_t, Whh_t, Why_t, bcat, nullptr, by, out);
  }
}
